// TestLaneATT_57054345560637
// MI455X (gfx1250) — matrix operand order repaired and checked by reading, not compiled, not run on hardware
//
#include <hip/hip_runtime.h>

typedef unsigned short u16;
typedef unsigned int   u32;
typedef __attribute__((ext_vector_type(8)))  u16    ushort8_t;
typedef __attribute__((ext_vector_type(16))) __bf16 v16bf;
typedef __attribute__((ext_vector_type(8)))  float  v8f;

#define N_ANCHORS 2784
#define CC 64
#define FH 90
#define FW 160
#define DD 5760          /* CC*FH */
#define NOFF 72
#define NHEAD 75         /* 2 cls + 73 reg */
#define PI_F 3.14159265358979323846f

__constant__ float c_left[6]   = {72.f,60.f,49.f,39.f,30.f,22.f};
__constant__ float c_right[6]  = {108.f,120.f,131.f,141.f,150.f,158.f};
__constant__ float c_bottom[15]= {165.f,150.f,141.f,131.f,120.f,108.f,100.f,90.f,80.f,72.f,60.f,49.f,39.f,30.f,15.f};

__device__ __forceinline__ u16 f2bf(float x) {
    u32 u = __builtin_bit_cast(u32, x);
    u32 r = u + 0x7fffu + ((u >> 16) & 1u);   // round-to-nearest-even
    return (u16)(r >> 16);
}
__device__ __forceinline__ float bf2f(u16 h) {
    u32 u = ((u32)h) << 16;
    return __builtin_bit_cast(float, u);
}

// Async 16-byte Global -> LDS copy (gfx1250 GLOBAL_LOAD_ASYNC_TO_LDS_B128,
// tracked by ASYNCcnt).  lds_off is the LDS byte address (low 32 bits of the
// generic pointer, see ISA 10.2).
__device__ __forceinline__ void async_copy16(u32 lds_off, const void* gptr) {
    asm volatile("global_load_async_to_lds_b128 %0, %1, off"
                 :: "v"(lds_off), "v"(gptr) : "memory");
}

// ---------------------------------------------------------------------------
// 1) Anchor geometry: sx, sy, xs[72] (pixel coords), cutx[90] (-1 = invalid)
// ---------------------------------------------------------------------------
__global__ void geom_kernel(float* __restrict__ sxA, float* __restrict__ syA,
                            float* __restrict__ xsA, int* __restrict__ cutxA) {
    int n = blockIdx.x * blockDim.x + threadIdx.x;
    if (n >= N_ANCHORS) return;
    float sx, sy, ang;
    if (n < 432)       { int si = n / 6,  ai = n % 6;          sx = 0.f; sy = 1.f - (float)si / 71.f;  ang = c_left[ai]; }
    else if (n < 2352) { int m = n - 432; int si = m / 15, ai = m % 15; sy = 1.f; sx = 1.f - (float)si / 127.f; ang = c_bottom[ai]; }
    else               { int m = n - 2352; int si = m / 6, ai = m % 6;  sx = 1.f; sy = 1.f - (float)si / 71.f;  ang = c_right[ai]; }
    float invt = 1.f / tanf(ang * (PI_F / 180.f));
    sxA[n] = sx; syA[n] = sy;
    #pragma unroll 4
    for (int i = 0; i < NOFF; ++i) {
        float ys = 1.f - (float)i / 71.f;
        xsA[n * NOFF + i] = (sx + (sy - ys) * invt) * 640.f;
    }
    for (int j = 0; j < FH; ++j) {
        float yc = 1.f - (float)j / 89.f;
        float xr = (sx + (sy - yc) * invt) * 640.f;
        float xi = rintf(xr * 0.25f);
        int h = 89 - j;   // flipped
        int v;
        if (xi < 0.f || xi > 160.f) v = -1;
        else {
            float cl = fminf(fmaxf(xi, 0.f), 159.f);
            v = (int)cl;
        }
        cutxA[n * FH + h] = v;
    }
}

// ---------------------------------------------------------------------------
// 2) 1x1 conv: f[b][o][pix] = sum_c feat[b][c][pix] * w[o][c] + bias[o]
// ---------------------------------------------------------------------------
__global__ void conv1x1_kernel(const float* __restrict__ feat, const float* __restrict__ w,
                               const float* __restrict__ bias, float* __restrict__ f, long total) {
    long t = (long)blockIdx.x * blockDim.x + threadIdx.x;
    if (t >= total) return;
    const long HW = (long)FH * FW;
    int pix = (int)(t % HW);
    long tmp = t / HW;
    int o = (int)(tmp % CC);
    int b = (int)(tmp / CC);
    const float* fin = feat + (long)b * CC * HW;
    float s = bias[o];
    #pragma unroll 8
    for (int c = 0; c < CC; ++c)
        s = fmaf(fin[(long)c * HW + pix], w[o * CC + c], s);
    f[t] = s;
}

// ---------------------------------------------------------------------------
// 3) fp32 -> bf16 convert
// ---------------------------------------------------------------------------
__global__ void cvt_bf16_kernel(const float* __restrict__ in, u16* __restrict__ out, long n) {
    long t = (long)blockIdx.x * blockDim.x + threadIdx.x;
    if (t < n) out[t] = f2bf(in[t]);
}

// ---------------------------------------------------------------------------
// 4) Anchor feature gather: F[b][n][c*90+h] = f[b][c][h][cutx[n][h]] (0 if invalid)
// ---------------------------------------------------------------------------
__global__ void gather_kernel(const float* __restrict__ f, const int* __restrict__ cutx,
                              u16* __restrict__ F, long total) {
    long t = (long)blockIdx.x * blockDim.x + threadIdx.x;
    if (t >= total) return;
    int e = (int)(t % DD);
    long tmp = t / DD;
    int n = (int)(tmp % N_ANCHORS);
    int b = (int)(tmp / N_ANCHORS);
    int c = e / FH, h = e % FH;
    int x = cutx[n * FH + h];
    float v = 0.f;
    if (x >= 0) v = f[((((long)b * CC + c) * FH) + h) * FW + x];
    F[t] = f2bf(v);
}

// ---------------------------------------------------------------------------
// 5/7) Double-buffered, async-pipelined bf16 WMMA GEMM.
//      C[M,N] = X[M,K] * op(W) + bias
//      BMODE 0: W stored [N][K] (NT) -> async Global->LDS for both tiles
//      BMODE 1: W stored [K][N] (NN) -> async for X, manual transpose for W
//      OMODE 0: fp32 out + bias[gn];  OMODE 1: bf16 out
//      OOB rows/cols are CLAMPED on load (uniform async issue count); the
//      resulting garbage LDS rows feed only output rows/cols that the store
//      guard masks (WMMA outputs depend only on their own A-row / B-col).
// ---------------------------------------------------------------------------
union FragU { v16bf v; ushort8_t h[2]; };

template<int BMODE, int OMODE>
__global__ __launch_bounds__(256)
void gemm_wmma_bf16(const u16* __restrict__ Xb, long xBatch, long ldx,
                    const u16* __restrict__ Wb, long wBatch, long ldw,
                    float* __restrict__ Cf, u16* __restrict__ Cb, long cBatch, long ldc,
                    const float* __restrict__ bias, int M, int N, int K) {
    constexpr int BM = 128, BN = 64, BK = 32;
    __shared__ u16 ldsX[2][BM * BK];
    __shared__ u16 ldsW[2][BN * BK];

    const int tid = threadIdx.x;
    const int m0 = blockIdx.x * BM;
    const int n0 = blockIdx.y * BN;
    const u16* Xp = Xb + (long)blockIdx.z * xBatch;
    const u16* Wp = Wb + (long)blockIdx.z * wBatch;

    const int lane = tid & 31, wv = tid >> 5;
    const int wm = (wv >> 1) * 32;       // 0,32,64,96
    const int wn = (wv & 1) * 32;        // 0,32
    const int half = lane >> 4;          // lane group
    const int l16 = lane & 15;
    const int kb = half * 8;             // A-frag K base per ISA layout

    // per-thread tile-load coordinates (fixed across K loop)
    const int xRow0 = tid >> 2;               // chunk 0 row (then +64)
    const int xKg   = (tid & 3) * 8;
    const int xGr0  = min(m0 + xRow0,      M - 1);
    const int xGr1  = min(m0 + xRow0 + 64, M - 1);
    const int wRowNT = tid >> 2;              // NT: one chunk per thread
    const int wKgNT  = (tid & 3) * 8;
    const int wGrNT  = min(n0 + wRowNT, N - 1);
    const int wKlNN  = tid >> 3;              // NN: k row within tile
    const int wN8NN  = (tid & 7) * 8;

    auto issue_stage = [&](int k0, int buf) {
        // X tile: 2 async b128 per thread
        async_copy16((u32)(size_t)&ldsX[buf][ xRow0       * BK + xKg],
                     Xp + (long)xGr0 * ldx + k0 + xKg);
        async_copy16((u32)(size_t)&ldsX[buf][(xRow0 + 64) * BK + xKg],
                     Xp + (long)xGr1 * ldx + k0 + xKg);
        if (BMODE == 0) {
            async_copy16((u32)(size_t)&ldsW[buf][wRowNT * BK + wKgNT],
                         Wp + (long)wGrNT * ldw + k0 + wKgNT);
        } else {
            int gk = k0 + wKlNN;
            int gn = n0 + wN8NN;
            ushort8_t val = (ushort8_t)0;
            if (gn + 8 <= N) {
                val = *(const ushort8_t*)(Wp + (long)gk * ldw + gn);
            } else {
                #pragma unroll
                for (int e = 0; e < 8; ++e) if (gn + e < N) val[e] = Wp[(long)gk * ldw + gn + e];
            }
            #pragma unroll
            for (int e = 0; e < 8; ++e) ldsW[buf][(wN8NN + e) * BK + wKlNN] = val[e];
        }
    };

    v8f acc[2][2] = {};
    const int nstages = K / BK;          // K % 32 == 0 for all uses

    issue_stage(0, 0);                   // prologue

    for (int s = 0; s < nstages; ++s) {
        const int buf = s & 1;
        if (s + 1 < nstages) {
            issue_stage((s + 1) * BK, buf ^ 1);
            // async loads complete in order: draining to the count just issued
            // guarantees stage s is fully in LDS.
            if (BMODE == 0) asm volatile("s_wait_asynccnt 0x3" ::: "memory");
            else            asm volatile("s_wait_asynccnt 0x2" ::: "memory");
        } else {
            asm volatile("s_wait_asynccnt 0x0" ::: "memory");
        }
        __syncthreads();

        // ---- build fragments (ISA 7.12.2 layouts) & 4x WMMA ----
        FragU fa[2], fb[2];
        #pragma unroll
        for (int ss = 0; ss < 2; ++ss) {
            int row = wm + ss * 16 + l16;
            int base = row * BK + kb;
            fa[ss].h[0] = *(const ushort8_t*)&ldsX[buf][base];
            fa[ss].h[1] = *(const ushort8_t*)&ldsX[buf][base + 16];
        }
        #pragma unroll
        for (int t2 = 0; t2 < 2; ++t2) {
            int col = wn + t2 * 16 + l16;
            int base = col * BK + kb;
            fb[t2].h[0] = *(const ushort8_t*)&ldsW[buf][base];
            fb[t2].h[1] = *(const ushort8_t*)&ldsW[buf][base + 16];
        }
        #pragma unroll
        for (int ss = 0; ss < 2; ++ss)
            #pragma unroll
            for (int t2 = 0; t2 < 2; ++t2)
                acc[ss][t2] = __builtin_amdgcn_wmma_f32_16x16x32_bf16(
                    false, fa[ss].v, false, fb[t2].v, (short)0, acc[ss][t2], false, false);
        __syncthreads();   // all waves done reading buf before stage s+2 overwrites it
    }

    // ---- store (C/D layout: VGPR r -> M=r / M=8+r for upper lanes) ----
    float* Cfp = (OMODE == 0) ? Cf + (long)blockIdx.z * cBatch : nullptr;
    u16*   Cbp = (OMODE == 1) ? Cb + (long)blockIdx.z * cBatch : nullptr;
    #pragma unroll
    for (int ss = 0; ss < 2; ++ss) {
        #pragma unroll
        for (int t2 = 0; t2 < 2; ++t2) {
            #pragma unroll
            for (int r = 0; r < 8; ++r) {
                int gm = m0 + wm + ss * 16 + half * 8 + r;
                int gn = n0 + wn + t2 * 16 + l16;
                if (gm < M && gn < N) {
                    float v = acc[ss][t2][r];
                    if (OMODE == 0) Cfp[(long)gm * ldc + gn] = v + (bias ? bias[gn] : 0.f);
                    else            Cbp[(long)gm * ldc + gn] = f2bf(v);
                }
            }
        }
    }
}

// ---------------------------------------------------------------------------
// 6) Softmax over 2783 scores + off-diagonal scatter into A[b][n][2784] (bf16)
// ---------------------------------------------------------------------------
__global__ __launch_bounds__(256)
void softmax_scatter_kernel(const float* __restrict__ S, u16* __restrict__ A) {
    __shared__ float red[256];
    const int n = blockIdx.x, b = blockIdx.y, tid = threadIdx.x;
    const float* row = S + ((long)b * N_ANCHORS + n) * N_ANCHORS;
    u16* arow = A + ((long)b * N_ANCHORS + n) * N_ANCHORS;
    const int KN = N_ANCHORS - 1;

    float m = -1e30f;
    for (int k = tid; k < KN; k += 256) m = fmaxf(m, row[k]);
    red[tid] = m; __syncthreads();
    for (int s = 128; s > 0; s >>= 1) { if (tid < s) red[tid] = fmaxf(red[tid], red[tid + s]); __syncthreads(); }
    float mx = red[0]; __syncthreads();

    float sum = 0.f;
    for (int k = tid; k < KN; k += 256) sum += __expf(row[k] - mx);
    red[tid] = sum; __syncthreads();
    for (int s = 128; s > 0; s >>= 1) { if (tid < s) red[tid] += red[tid + s]; __syncthreads(); }
    float inv = 1.f / red[0]; __syncthreads();

    for (int k = tid; k < KN; k += 256) {
        int j = k + (k >= n ? 1 : 0);
        arow[j] = f2bf(__expf(row[k] - mx) * inv);
    }
    if (tid == 0) arow[n] = 0;   // bf16 zero
}

// ---------------------------------------------------------------------------
// 8) Heads: heads[b][n][o] = dot(cat[b][n], w_o) + bias_o,  cat = [att_feat | F]
// ---------------------------------------------------------------------------
__global__ __launch_bounds__(256)
void heads_kernel(const u16* __restrict__ attF, const u16* __restrict__ F,
                  const float* __restrict__ cls_w, const float* __restrict__ cls_b,
                  const float* __restrict__ reg_w, const float* __restrict__ reg_b,
                  float* __restrict__ heads) {
    __shared__ float catf[2 * DD];
    __shared__ float red[256];
    const int n = blockIdx.x, b = blockIdx.y, tid = threadIdx.x;
    const u16* ar = attF + ((long)b * N_ANCHORS + n) * DD;
    const u16* fr = F    + ((long)b * N_ANCHORS + n) * DD;
    for (int i = tid; i < DD; i += 256) { catf[i] = bf2f(ar[i]); catf[DD + i] = bf2f(fr[i]); }
    __syncthreads();
    for (int o = 0; o < NHEAD; ++o) {
        const float* w = (o < 2) ? (cls_w + (long)o * 2 * DD) : (reg_w + (long)(o - 2) * 2 * DD);
        float s = 0.f;
        for (int i = tid; i < 2 * DD; i += 256) s = fmaf(catf[i], w[i], s);
        red[tid] = s; __syncthreads();
        for (int st = 128; st > 0; st >>= 1) { if (tid < st) red[tid] += red[tid + st]; __syncthreads(); }
        if (tid == 0)
            heads[((long)b * N_ANCHORS + n) * NHEAD + o] = red[0] + (o < 2 ? cls_b[o] : reg_b[o - 2]);
        __syncthreads();
    }
}

// ---------------------------------------------------------------------------
// 9) Assemble props [B][2784][77]
// ---------------------------------------------------------------------------
__global__ void assemble_kernel(const float* __restrict__ heads, const float* __restrict__ sxA,
                                const float* __restrict__ syA, const float* __restrict__ xsA,
                                float* __restrict__ out, int B) {
    int t = blockIdx.x * blockDim.x + threadIdx.x;
    if (t >= B * N_ANCHORS) return;
    int n = t % N_ANCHORS;
    const float* h = heads + (long)t * NHEAD;
    float* o = out + (long)t * 77;
    o[0] = h[0]; o[1] = h[1];
    o[2] = 1.f - syA[n]; o[3] = sxA[n];
    o[4] = h[2];                          // base col4 == 0
    #pragma unroll 4
    for (int i = 0; i < NOFF; ++i) o[5 + i] = xsA[n * NOFF + i] + h[3 + i];
}

// ---------------------------------------------------------------------------
extern "C" void kernel_launch(void* const* d_in, const int* in_sizes, int n_in,
                              void* d_out, int out_size, void* d_ws, size_t ws_size,
                              hipStream_t stream) {
    const float* feat   = (const float*)d_in[0];
    const float* conv1w = (const float*)d_in[1];
    const float* conv1b = (const float*)d_in[2];
    const float* attnw  = (const float*)d_in[3];
    const float* attnb  = (const float*)d_in[4];
    const float* clsw   = (const float*)d_in[5];
    const float* clsb   = (const float*)d_in[6];
    const float* regw   = (const float*)d_in[7];
    const float* regb   = (const float*)d_in[8];
    float* outp = (float*)d_out;

    const int B = in_sizes[0] / (CC * FH * FW);

    // workspace carve-out (256B aligned)
    size_t off = 0;
    auto alloc = [&](size_t bytes) -> char* {
        size_t p = (off + 255) & ~(size_t)255;
        off = p + bytes;
        return (char*)d_ws + p;
    };
    float* g_sx   = (float*)alloc((size_t)N_ANCHORS * 4);
    float* g_sy   = (float*)alloc((size_t)N_ANCHORS * 4);
    float* g_xs   = (float*)alloc((size_t)N_ANCHORS * NOFF * 4);
    int*   g_cutx = (int*)  alloc((size_t)N_ANCHORS * FH * 4);
    float* g_f    = (float*)alloc((size_t)B * CC * FH * FW * 4);
    u16*   g_F    = (u16*)  alloc((size_t)B * N_ANCHORS * DD * 2);
    u16*   g_W    = (u16*)  alloc((size_t)(N_ANCHORS - 1) * DD * 2);
    size_t scoresB = (size_t)B * N_ANCHORS * N_ANCHORS * 4;
    size_t attB    = (size_t)B * N_ANCHORS * DD * 2;
    char*  g_union = alloc(scoresB > attB ? scoresB : attB);   // scores dies before att_feat is written
    float* g_scores = (float*)g_union;
    u16*   g_attF   = (u16*)g_union;
    u16*   g_A    = (u16*)  alloc((size_t)B * N_ANCHORS * N_ANCHORS * 2);
    float* g_heads= (float*)alloc((size_t)B * N_ANCHORS * NHEAD * 4);

    // 1) geometry
    geom_kernel<<<(N_ANCHORS + 255) / 256, 256, 0, stream>>>(g_sx, g_sy, g_xs, g_cutx);
    // 2) 1x1 conv
    long convTot = (long)B * CC * FH * FW;
    conv1x1_kernel<<<(int)((convTot + 255) / 256), 256, 0, stream>>>(feat, conv1w, conv1b, g_f, convTot);
    // 3) attn_w -> bf16
    long wTot = (long)(N_ANCHORS - 1) * DD;
    cvt_bf16_kernel<<<(int)((wTot + 255) / 256), 256, 0, stream>>>(attnw, g_W, wTot);
    // 4) gather -> F (bf16)
    long gTot = (long)B * N_ANCHORS * DD;
    gather_kernel<<<(int)((gTot + 255) / 256), 256, 0, stream>>>(g_f, g_cutx, g_F, gTot);
    // 5) scores = F * attn_w^T + attn_b   (M=2784, N=2783, K=5760)
    {
        dim3 grid((N_ANCHORS + 127) / 128, (N_ANCHORS - 1 + 63) / 64, B);
        gemm_wmma_bf16<0, 0><<<grid, 256, 0, stream>>>(
            g_F, (long)N_ANCHORS * DD, DD,
            g_W, 0, DD,
            g_scores, nullptr, (long)N_ANCHORS * N_ANCHORS, N_ANCHORS,
            attnb, N_ANCHORS, N_ANCHORS - 1, DD);
    }
    // 6) softmax + scatter -> A (bf16, diag 0)
    softmax_scatter_kernel<<<dim3(N_ANCHORS, B), 256, 0, stream>>>(g_scores, g_A);
    // 7) att_feat = A * F   (M=2784, N=5760, K=2784) — overwrites scores region
    {
        dim3 grid((N_ANCHORS + 127) / 128, (DD + 63) / 64, B);
        gemm_wmma_bf16<1, 1><<<grid, 256, 0, stream>>>(
            g_A, (long)N_ANCHORS * N_ANCHORS, N_ANCHORS,
            g_F, (long)N_ANCHORS * DD, DD,
            nullptr, g_attF, (long)N_ANCHORS * DD, DD,
            nullptr, N_ANCHORS, DD, N_ANCHORS);
    }
    // 8) heads
    heads_kernel<<<dim3(N_ANCHORS, B), 256, 0, stream>>>(g_attF, g_F, clsw, clsb, regw, regb, g_heads);
    // 9) assemble props
    assemble_kernel<<<(B * N_ANCHORS + 255) / 256, 256, 0, stream>>>(g_heads, g_sx, g_sy, g_xs, outp, B);
}
